// Policy_76751065579639
// MI455X (gfx1250) — matrix operand order repaired and checked by reading, not compiled, not run on hardware
//
#include <hip/hip_runtime.h>
#include <hip/hip_bf16.h>

typedef __attribute__((ext_vector_type(16))) _Float16 v16h;
typedef __attribute__((ext_vector_type(8)))  _Float16 v8h;
typedef __attribute__((ext_vector_type(8)))  float    v8f;

#define BSZ   32768
#define MTOK  200
#define NL    22
#define CHN   128

// ---- workspace layout (in f16 elements) ----
// B1   : [8 tiles][25 chunks][32 lanes][16] = 102400   (K = (dx*5+dy)*32 + c, c padded 22->32)
// B2   : [8][36][32][16]                    = 147456
// BFC  : [16][4][32][16]                    = 32768
// BSELF: [16][1][32][16]                    = 8192
// BHEAD: [2][16][32][16]                    = 16384
#define OFF_B2    102400
#define OFF_BFC   249856
#define OFF_BSELF 282624
#define OFF_BHEAD 290816

// ---- LDS layout (f16 elements) ----
// box2 : 16 samples * 121 pixels * 32 ch    = 61952   (channel-last, zero padded)
// a2   : 16 * 1152                          = 18432
// h2   : 16 * 128                           =  2048
// hid  : 16 * 512                           =  8192
// maxv : 32 floats at tail
#define SSTRIDE  3872            // 121*32, per-sample stride in box2
#define L_A2     61952
#define L_H2     80384
#define L_HID    82432
#define L_MAXV   90624
#define SMEM_BYTES (90624*2 + 128)

// ======================= weight prep kernels =========================
// B-fragment layout: dst[((t*CHUNKS + k)*32 + lane)*16 + s]
// column N = t*16 + (lane&15); K = k*32 + (lane>>4)*16 + s
__global__ void prep_b1(const float* __restrict__ w, _Float16* __restrict__ dst) {
    int i = blockIdx.x * 256 + threadIdx.x;
    if (i >= 8 * 25 * 512) return;
    int s = i & 15, lane = (i >> 4) & 31, k = (i >> 9) % 25, t = (i >> 9) / 25;
    int N = t * 16 + (lane & 15);
    int c = (lane >> 4) * 16 + s;          // channel within padded-32 chunk
    int dx = k / 5, dy = k - dx * 5;
    float v = (c < NL) ? w[((N * NL + c) * 5 + dx) * 5 + dy] : 0.f;
    dst[i] = (_Float16)v;
}

__global__ void prep_b2(const float* __restrict__ w, _Float16* __restrict__ dst) {
    int i = blockIdx.x * 256 + threadIdx.x;
    if (i >= 8 * 36 * 512) return;
    int s = i & 15, lane = (i >> 4) & 31, k = (i >> 9) % 36, t = (i >> 9) / 36;
    int N = t * 16 + (lane & 15);
    int kk = k * 32 + (lane >> 4) * 16 + s;           // kk < 1152 always
    int c = kk / 9, rem = kk - c * 9, dx = rem / 3, dy = rem - dx * 3;
    dst[i] = (_Float16)w[((N * CHN + c) * 3 + dx) * 3 + dy];
}

__global__ void prep_bfc(const float* __restrict__ w, _Float16* __restrict__ dst) {
    int i = blockIdx.x * 256 + threadIdx.x;
    if (i >= 16 * 4 * 512) return;
    int s = i & 15, lane = (i >> 4) & 31, k = (i >> 9) % 4, t = (i >> 9) / 4;
    int N = t * 16 + (lane & 15);
    int kk = k * 32 + (lane >> 4) * 16 + s;           // kk < 128 always
    dst[i] = (_Float16)w[kk * 256 + N];
}

__global__ void prep_bself(const float* __restrict__ w, _Float16* __restrict__ dst) {
    int i = blockIdx.x * 256 + threadIdx.x;
    if (i >= 16 * 512) return;
    int s = i & 15, lane = (i >> 4) & 31, t = i >> 9;
    int N = t * 16 + (lane & 15);
    int kk = (lane >> 4) * 16 + s;
    float v = (kk < NL) ? w[kk * 256 + N] : 0.f;
    dst[i] = (_Float16)v;
}

__global__ void prep_bhead(const float* __restrict__ wa0, const float* __restrict__ wa1,
                           const float* __restrict__ wv, _Float16* __restrict__ dst) {
    int i = blockIdx.x * 256 + threadIdx.x;
    if (i >= 2 * 16 * 512) return;
    int s = i & 15, lane = (i >> 4) & 31, k = (i >> 9) % 16, t = (i >> 9) / 16;
    int N = t * 16 + (lane & 15);
    int kk = k * 32 + (lane >> 4) * 16 + s;           // kk < 512
    float v = 0.f;
    if      (N < 9)   v = wa0[kk * 9 + N];
    else if (N < 19)  v = wa1[kk * 10 + (N - 9)];
    else if (N == 19) v = wv[kk];
    dst[i] = (_Float16)v;
}

// ======================= fragment helpers =========================
__device__ __forceinline__ v16h load_a(const _Float16* rowp, int half) {
    // contiguous k order used here (B packed to match): lane<16: K 0-15; lane>=16: K 16-31
    return *(const v16h*)(rowp + half * 16);
}

__device__ __forceinline__ v8f wmma16(v16h a, v16h b, v8f c) {
    return __builtin_amdgcn_wmma_f32_16x16x32_f16(false, a, false, b, (short)0, c, false, false);
}

// ======================= fused forward kernel =========================
__global__ void __launch_bounds__(256)
fused_policy(const int* __restrict__ obs, const float* __restrict__ max_vec,
             const float* __restrict__ bc1, const float* __restrict__ bc2,
             const float* __restrict__ bfc, const float* __restrict__ bself,
             const float* __restrict__ ba0, const float* __restrict__ ba1,
             const float* __restrict__ bv,
             const _Float16* __restrict__ wsB1, const _Float16* __restrict__ wsB2,
             const _Float16* __restrict__ wsBfc, const _Float16* __restrict__ wsBself,
             const _Float16* __restrict__ wsBhead,
             float* __restrict__ out0, float* __restrict__ out1,
             float* __restrict__ outv, float* __restrict__ outh) {
    extern __shared__ _Float16 smem[];
    _Float16* box2 = smem;                 // [16][11*11][32] channel-last
    _Float16* a2   = smem + L_A2;          // [16][1152] conv2 im2col
    _Float16* h2   = smem + L_H2;          // [16][128]
    _Float16* hid  = smem + L_HID;         // [16][512]
    float*    mv   = (float*)(smem + L_MAXV);

    const int tid   = threadIdx.x;
    const int lane  = tid & 31;
    const int w     = tid >> 5;       // wave id 0..7
    const int row   = lane & 15;      // sample row for A-frags, N col for D
    const int half  = lane >> 4;
    const int sbase = blockIdx.x * 16;

    // -- clear box2 (incl. channel padding), stage max_vec --
    { unsigned* p = (unsigned*)box2;
      for (int i = tid; i < 30976; i += 256) p[i] = 0u; }
    if (tid < NL) mv[tid] = max_vec[tid];
    __syncthreads();

    // -- token scatter into box2 (f16, pre-divided by max_vec) --
    const int* ob = obs + (size_t)sbase * MTOK * 3;
    for (int i = tid; i < 16 * MTOK; i += 256) {
        int o0 = ob[i * 3 + 0]; if (o0 == 255) o0 = 0;
        int o1 = ob[i * 3 + 1]; if (o1 == 255) o1 = 0;
        int o2 = ob[i * 3 + 2]; if (o2 == 255) o2 = 0;
        int x = (o0 >> 4) & 15, y = o0 & 15;
        if (x < 11 && y < 11 && o1 < NL) {
            int s = i / MTOK;
            box2[s * SSTRIDE + (x * 11 + y) * 32 + o1] =
                (_Float16)((float)o2 / mv[o1]);
        }
    }
    __syncthreads();

    // ============ conv1: 9 positions, K = 25 chunks of 32 (channel-last) ============
    const float bias1 = bc1[w * 16 + row];
    // preload this wave's 25 B fragments (reused across all 9 positions)
    v16h b1f[25];
    #pragma unroll
    for (int q = 0; q < 25; ++q)
        b1f[q] = *(const v16h*)(wsB1 + ((size_t)(w * 25 + q) * 32 + lane) * 16);

    const _Float16* arow = box2 + row * SSTRIDE;
    for (int p = 0; p < 9; ++p) {
        const int ix0 = (p / 3) * 3, iy0 = (p % 3) * 3;
        v8f acc = {};
        #pragma unroll
        for (int dx = 0; dx < 5; ++dx) {
            v16h af[5];
            #pragma unroll
            for (int dy = 0; dy < 5; ++dy)
                af[dy] = load_a(arow + ((ix0 + dx) * 11 + iy0 + dy) * 32, half);
            #pragma unroll
            for (int dy = 0; dy < 5; ++dy)
                acc = wmma16(af[dy], b1f[dx * 5 + dy], acc);
        }
        // bias + relu, scatter into conv2 im2col buffer: K = ch*9 + p
        #pragma unroll
        for (int r = 0; r < 8; ++r) {
            float v = acc[r] + bias1; v = v > 0.f ? v : 0.f;
            a2[(half * 8 + r) * 1152 + (w * 16 + row) * 9 + p] = (_Float16)v;
        }
    }
    __syncthreads();

    // ================= conv2: [16x1152]x[1152x16] per wave =================
    {
        const float bias2 = bc2[w * 16 + row];
        v8f acc = {};
        const _Float16* ar = a2 + row * 1152;
        for (int kb = 0; kb < 36; kb += 6) {
            v16h af[6];
            #pragma unroll
            for (int j = 0; j < 6; ++j)
                af[j] = load_a(ar + (kb + j) * 32, half);
            #pragma unroll
            for (int j = 0; j < 6; ++j) {
                v16h b = *(const v16h*)(wsB2 + ((size_t)(w * 36 + kb + j) * 32 + lane) * 16);
                acc = wmma16(af[j], b, acc);
            }
        }
        #pragma unroll
        for (int r = 0; r < 8; ++r) {
            float v = acc[r] + bias2; v = v > 0.f ? v : 0.f;
            h2[(half * 8 + r) * 128 + w * 16 + row] = (_Float16)v;
        }
    }
    __syncthreads();

    // ================= FC + self GEMMs -> hidden [16x512] =================
    float* outh_blk = outh + (size_t)sbase * 512;
    #pragma unroll
    for (int tt = 0; tt < 2; ++tt) {
        const int t = w * 2 + tt;
        const int col = t * 16 + row;
        // cnn_feat tile: K=128
        {
            v8f acc = {};
            const _Float16* ar = h2 + row * 128;
            v16h af[4];
            #pragma unroll
            for (int k = 0; k < 4; ++k)
                af[k] = load_a(ar + k * 32, half);
            #pragma unroll
            for (int k = 0; k < 4; ++k) {
                v16h b = *(const v16h*)(wsBfc + ((size_t)(t * 4 + k) * 32 + lane) * 16);
                acc = wmma16(af[k], b, acc);
            }
            const float bias = bfc[col];
            #pragma unroll
            for (int r = 0; r < 8; ++r) {
                float v = acc[r] + bias; v = v > 0.f ? v : 0.f;
                int s = half * 8 + r;
                hid[s * 512 + 256 + col] = (_Float16)v;
                outh_blk[s * 512 + 256 + col] = v;
            }
        }
        // self_feat tile: K=32 (22 channels, padded zeros) — A direct from box center
        {
            v8f acc = {};
            v16h a = load_a(box2 + row * SSTRIDE + 60 * 32, half);   // pixel (5,5)
            v16h b = *(const v16h*)(wsBself + ((size_t)(t * 32) + lane) * 16);
            acc = wmma16(a, b, acc);
            const float bias = bself[col];
            #pragma unroll
            for (int r = 0; r < 8; ++r) {
                float v = acc[r] + bias; v = v > 0.f ? v : 0.f;
                int s = half * 8 + r;
                hid[s * 512 + col] = (_Float16)v;
                outh_blk[s * 512 + col] = v;
            }
        }
    }
    __syncthreads();

    // ================= heads: [16x512]x[512x32] (20 real cols) =================
    if (w < 2) {
        v8f acc = {};
        const _Float16* ar = hid + row * 512;
        for (int kb = 0; kb < 16; kb += 4) {
            v16h af[4];
            #pragma unroll
            for (int j = 0; j < 4; ++j)
                af[j] = load_a(ar + (kb + j) * 32, half);
            #pragma unroll
            for (int j = 0; j < 4; ++j) {
                v16h b = *(const v16h*)(wsBhead + ((size_t)(w * 16 + kb + j) * 32 + lane) * 16);
                acc = wmma16(af[j], b, acc);
            }
        }
        const int col = w * 16 + row;
        #pragma unroll
        for (int r = 0; r < 8; ++r) {
            int sg = sbase + half * 8 + r;
            float v = acc[r];
            if (col < 9)        out0[sg * 9 + col]        = v + ba0[col];
            else if (col < 19)  out1[sg * 10 + (col - 9)] = v + ba1[col - 9];
            else if (col == 19) outv[sg]                  = v + bv[0];
        }
    }
}

// ======================= host launch =========================
extern "C" void kernel_launch(void* const* d_in, const int* in_sizes, int n_in,
                              void* d_out, int out_size, void* d_ws, size_t ws_size,
                              hipStream_t stream) {
    (void)in_sizes; (void)n_in; (void)out_size; (void)ws_size;
    const int*   obs    = (const int*)d_in[0];
    const float* maxv   = (const float*)d_in[1];
    const float* w1     = (const float*)d_in[2];
    const float* b1     = (const float*)d_in[3];
    const float* w2     = (const float*)d_in[4];
    const float* b2     = (const float*)d_in[5];
    const float* wfc    = (const float*)d_in[6];
    const float* bfc    = (const float*)d_in[7];
    const float* wself  = (const float*)d_in[8];
    const float* bself  = (const float*)d_in[9];
    const float* wa0    = (const float*)d_in[10];
    const float* ba0    = (const float*)d_in[11];
    const float* wa1    = (const float*)d_in[12];
    const float* ba1    = (const float*)d_in[13];
    const float* wv     = (const float*)d_in[14];
    const float* bv     = (const float*)d_in[15];

    _Float16* ws     = (_Float16*)d_ws;
    _Float16* B1     = ws;
    _Float16* B2     = ws + OFF_B2;
    _Float16* Bfc    = ws + OFF_BFC;
    _Float16* Bself  = ws + OFF_BSELF;
    _Float16* Bhead  = ws + OFF_BHEAD;

    prep_b1   <<<400, 256, 0, stream>>>(w1, B1);
    prep_b2   <<<576, 256, 0, stream>>>(w2, B2);
    prep_bfc  <<<128, 256, 0, stream>>>(wfc, Bfc);
    prep_bself<<<32,  256, 0, stream>>>(wself, Bself);
    prep_bhead<<<64,  256, 0, stream>>>(wa0, wa1, wv, Bhead);

    float* out0 = (float*)d_out;
    float* out1 = out0 + (size_t)BSZ * 9;
    float* outv = out1 + (size_t)BSZ * 10;
    float* outh = outv + (size_t)BSZ;

    fused_policy<<<BSZ / 16, 256, SMEM_BYTES, stream>>>(
        obs, maxv, b1, b2, bfc, bself, ba0, ba1, bv,
        B1, B2, Bfc, Bself, Bhead, out0, out1, outv, outh);
}
